// GNN_LCG_84370337563247
// MI455X (gfx1250) — matrix operand order repaired and checked by reading, not compiled, not run on hardware
//
#include <hip/hip_runtime.h>

typedef unsigned short u16;
typedef unsigned int   u32;
typedef __attribute__((ext_vector_type(16))) __bf16 v16bf;
typedef __attribute__((ext_vector_type(8)))  float  v8f;

#define TS 64   // rows per block for all GEMM-style kernels

union Frag16 {
  v16bf v;
  u16   s[16];
  uint4 q[2];
};

__device__ __forceinline__ u16 f2bf(float f) {
  u32 u = __float_as_uint(f);
  u32 r = u + 0x7fffu + ((u >> 16) & 1u);   // round-to-nearest-even
  return (u16)(r >> 16);
}

__device__ __forceinline__ void atomAddF(float* p, float v) {
  (void)__hip_atomic_fetch_add(p, v, __ATOMIC_RELAXED, __HIP_MEMORY_SCOPE_AGENT);
}

// ---------------------------------------------------------------------------
// Weight pre-pack: W (K x 128, row-major f32) -> bf16 B-fragments.
// Fragment index = nb*KB + kb. Per lane: n = lane&15 (+nothing), hi = lane>>4,
// half j holds W[kb*32 + (j<8 ? 8*hi + j : 16 + 8*hi + j - 8)][nb*16 + n].   (ISA 05_wmma B layout)
// ---------------------------------------------------------------------------
__global__ void pack_w_kernel(const float* __restrict__ W, u16* __restrict__ Wp, int KB) {
  int frag = blockIdx.x;                 // 0 .. 8*KB-1
  int nb = frag / KB;
  int kb = frag - nb * KB;
  int lane = threadIdx.x;
  int hi = lane >> 4, nl = lane & 15;
  for (int j = 0; j < 16; ++j) {
    int k = kb * 32 + (j < 8 ? 8 * hi + j : 16 + 8 * hi + (j - 8));
    Wp[(size_t)frag * 512 + lane * 16 + j] = f2bf(W[(size_t)k * 128 + nb * 16 + nl]);
  }
}

// ---------------------------------------------------------------------------
// Degree + normalization
// ---------------------------------------------------------------------------
__global__ void deg_kernel(const int* __restrict__ li, const int* __restrict__ ci,
                           float* ldeg, float* cdeg, int n) {
  int i = blockIdx.x * blockDim.x + threadIdx.x;
  if (i < n) {
    atomAddF(&ldeg[li[i]], 1.0f);
    atomAddF(&cdeg[ci[i]], 1.0f);
  }
}

__global__ void norm_kernel(const int* __restrict__ li, const int* __restrict__ ci,
                            const float* __restrict__ ldeg, const float* __restrict__ cdeg,
                            float* __restrict__ invn, int n) {
  int i = blockIdx.x * blockDim.x + threadIdx.x;
  if (i < n) invn[i] = 1.0f / (sqrtf(ldeg[li[i]]) * sqrtf(cdeg[ci[i]]));
}

__global__ void init_emb_kernel(const float* __restrict__ row, float* __restrict__ dst, int total) {
  int i = blockIdx.x * blockDim.x + threadIdx.x;
  if (i < total) dst[i] = row[i & 127];
}

// ---------------------------------------------------------------------------
// Fused 2-layer MLP: Out = relu(X @ W1 + b1) @ W2 + b2, rows = gridDim.x*64.
// rowSwap=1 reads row^1 (literal pair swap). All matmuls via WMMA bf16.
// ---------------------------------------------------------------------------
__global__ __launch_bounds__(128) void mlp2_kernel(
    const float* __restrict__ X, int rowSwap,
    const u16* __restrict__ W1, const float* __restrict__ B1,
    const u16* __restrict__ W2, const float* __restrict__ B2,
    float* __restrict__ Out)
{
  constexpr int S = 136;   // padded row stride in halfs (bank-shift, 16B aligned)
  __shared__ __align__(16) u16 ldsX[TS * S];
  __shared__ __align__(16) u16 ldsH[TS * S];

  const int tid  = threadIdx.x;
  const int base = blockIdx.x * TS;

  for (int idx = tid; idx < TS * 128; idx += 128) {
    int r = idx >> 7, c = idx & 127;
    ldsX[r * S + c] = f2bf(X[(size_t)(base + (r ^ rowSwap)) * 128 + c]);
  }
  __syncthreads();

  const int lane = tid & 31;
  const int w    = tid >> 5;
  const int hi   = lane >> 4;
  const int nl   = lane & 15;

  // A fragments for this wave's 16-row strip (16-bit A layout, ISA 05_wmma)
  const u16* xrow = ldsX + (w * 16 + nl) * S;
  Frag16 a[4];
#pragma unroll
  for (int kb = 0; kb < 4; ++kb) {
    a[kb].q[0] = *(const uint4*)(xrow + kb * 32 + 8 * hi);
    a[kb].q[1] = *(const uint4*)(xrow + kb * 32 + 16 + 8 * hi);
  }

  // ---- layer 1 ----
#pragma unroll
  for (int nb = 0; nb < 8; ++nb) {
    float bb = B1[nb * 16 + nl];
    v8f acc;
#pragma unroll
    for (int r = 0; r < 8; ++r) acc[r] = bb;
#pragma unroll
    for (int kb = 0; kb < 4; ++kb) {
      Frag16 b;
      const uint4* bp = ((const uint4*)W1) + (size_t)(nb * 4 + kb) * 64 + lane * 2;
      b.q[0] = bp[0]; b.q[1] = bp[1];
      acc = __builtin_amdgcn_wmma_f32_16x16x32_bf16(false, a[kb].v, false, b.v,
                                                    (short)0, acc, false, false);
    }
#pragma unroll
    for (int r = 0; r < 8; ++r) {
      float v0 = fmaxf(acc[r], 0.0f);
      ldsH[(w * 16 + r + 8 * hi) * S + nb * 16 + nl] = f2bf(v0);
    }
  }
  __syncthreads();

  // ---- layer 2 ----
  const u16* hrow = ldsH + (w * 16 + nl) * S;
  Frag16 h[4];
#pragma unroll
  for (int kb = 0; kb < 4; ++kb) {
    h[kb].q[0] = *(const uint4*)(hrow + kb * 32 + 8 * hi);
    h[kb].q[1] = *(const uint4*)(hrow + kb * 32 + 16 + 8 * hi);
  }

#pragma unroll
  for (int nb = 0; nb < 8; ++nb) {
    float bb = B2[nb * 16 + nl];
    v8f acc;
#pragma unroll
    for (int r = 0; r < 8; ++r) acc[r] = bb;
#pragma unroll
    for (int kb = 0; kb < 4; ++kb) {
      Frag16 b;
      const uint4* bp = ((const uint4*)W2) + (size_t)(nb * 4 + kb) * 64 + lane * 2;
      b.q[0] = bp[0]; b.q[1] = bp[1];
      acc = __builtin_amdgcn_wmma_f32_16x16x32_bf16(false, h[kb].v, false, b.v,
                                                    (short)0, acc, false, false);
    }
#pragma unroll
    for (int r = 0; r < 8; ++r)
      Out[(size_t)(base + w * 16 + r + 8 * hi) * 128 + nb * 16 + nl] = acc[r];
  }
}

// ---------------------------------------------------------------------------
// Single-layer update GEMM: Out = concat(S0[,S1[,S2]]) @ W + b, in-place safe
// (tile fully staged to LDS before any write; rows are block-private).
// ---------------------------------------------------------------------------
template <int NSRC>
__global__ __launch_bounds__(128) void update_kernel(
    const float* S0, const float* S1, const float* S2,
    const u16* __restrict__ Wp, const float* __restrict__ Bias,
    float* Out)
{
  constexpr int K  = NSRC * 128;
  constexpr int KB = K / 32;
  constexpr int S  = K + 8;
  __shared__ __align__(16) u16 ldsX[TS * S];

  const int tid  = threadIdx.x;
  const int base = blockIdx.x * TS;

  for (int idx = tid; idx < TS * K; idx += 128) {
    int r = idx / K, c = idx - r * K;
    const float* src = (c < 128) ? S0 : ((NSRC >= 3 && c >= 256) ? S2 : S1);
    int cc = c & 127;
    ldsX[r * S + c] = f2bf(src[(size_t)(base + r) * 128 + cc]);
  }
  __syncthreads();

  const int lane = tid & 31, w = tid >> 5;
  const int hi = lane >> 4, nl = lane & 15;
  const u16* xrow = ldsX + (w * 16 + nl) * S;

#pragma unroll
  for (int nb = 0; nb < 8; ++nb) {
    float bb = Bias[nb * 16 + nl];
    v8f acc;
#pragma unroll
    for (int r = 0; r < 8; ++r) acc[r] = bb;
#pragma unroll
    for (int kb = 0; kb < KB; ++kb) {
      Frag16 a;
      a.q[0] = *(const uint4*)(xrow + kb * 32 + 8 * hi);
      a.q[1] = *(const uint4*)(xrow + kb * 32 + 16 + 8 * hi);
      Frag16 b;
      const uint4* bp = ((const uint4*)Wp) + (size_t)(nb * KB + kb) * 64 + lane * 2;
      b.q[0] = bp[0]; b.q[1] = bp[1];
      acc = __builtin_amdgcn_wmma_f32_16x16x32_bf16(false, a.v, false, b.v,
                                                    (short)0, acc, false, false);
    }
#pragma unroll
    for (int r = 0; r < 8; ++r)
      Out[(size_t)(base + w * 16 + r + 8 * hi) * 128 + nb * 16 + nl] = acc[r];
  }
}

// ---------------------------------------------------------------------------
// Edge scatter: Aggr[dst[e]] += Msg[src[e]] * norm[e].  One wave per edge,
// float4 per lane, native f32 atomics (L2-resident working set).
// ---------------------------------------------------------------------------
__global__ __launch_bounds__(256) void scatter_kernel(
    const float* __restrict__ Msg, const int* __restrict__ SrcIdx,
    const int* __restrict__ DstIdx, const float* __restrict__ Norm,
    float* __restrict__ Aggr, int nEdges)
{
  int e = blockIdx.x * 8 + (threadIdx.x >> 5);
  if (e >= nEdges) return;
  int lane = threadIdx.x & 31;
  int sr = SrcIdx[e], dr = DstIdx[e];
  float wgt = Norm[e];
  float4 v = ((const float4*)(Msg + (size_t)sr * 128))[lane];
  float* dst = Aggr + (size_t)dr * 128 + lane * 4;
  atomAddF(dst + 0, v.x * wgt);
  atomAddF(dst + 1, v.y * wgt);
  atomAddF(dst + 2, v.z * wgt);
  atomAddF(dst + 3, v.w * wgt);
}

// ---------------------------------------------------------------------------
extern "C" void kernel_launch(void* const* d_in, const int* in_sizes, int n_in,
                              void* d_out, int out_size, void* d_ws, size_t ws_size,
                              hipStream_t stream)
{
  const int*   l_idx  = (const int*)d_in[0];
  const int*   c_idx  = (const int*)d_in[1];
  const float* l_init = (const float*)d_in[2];
  const float* c_init = (const float*)d_in[3];
  const float* w_l2c1 = (const float*)d_in[4];  const float* b_l2c1 = (const float*)d_in[5];
  const float* w_l2c2 = (const float*)d_in[6];  const float* b_l2c2 = (const float*)d_in[7];
  const float* w_c2l1 = (const float*)d_in[8];  const float* b_c2l1 = (const float*)d_in[9];
  const float* w_c2l2 = (const float*)d_in[10]; const float* b_c2l2 = (const float*)d_in[11];
  const float* w_l2l1 = (const float*)d_in[12]; const float* b_l2l1 = (const float*)d_in[13];
  const float* w_l2l2 = (const float*)d_in[14]; const float* b_l2l2 = (const float*)d_in[15];
  const float* w_cu   = (const float*)d_in[16]; const float* b_cu   = (const float*)d_in[17];
  const float* w_lu   = (const float*)d_in[18]; const float* b_lu   = (const float*)d_in[19];

  const int E = in_sizes[0];
  const int L = out_size / 128;   // 65536
  const int C = L / 2;            // 32768 per reference setup

  float* l_emb = (float*)d_out;   // working buffer == output

  unsigned char* p = (unsigned char*)d_ws;
  auto alloc = [&](size_t bytes) -> void* {
    void* r = (void*)p;
    p += (bytes + 255) & ~(size_t)255;
    return r;
  };
  float* c_emb   = (float*)alloc((size_t)C * 128 * 4);
  float* msg_l2c = (float*)alloc((size_t)L * 128 * 4);
  float* msg_c2l = (float*)alloc((size_t)C * 128 * 4);
  float* l2lmsg  = (float*)alloc((size_t)L * 128 * 4);
  float* aggr_c  = (float*)alloc((size_t)C * 128 * 4);
  float* aggr_l  = (float*)alloc((size_t)L * 128 * 4);
  float* ldeg    = (float*)alloc((size_t)L * 4);
  float* cdeg    = (float*)alloc((size_t)C * 4);
  float* invn    = (float*)alloc((size_t)E * 4);
  u16* p_l2c1 = (u16*)alloc(128 * 128 * 2);
  u16* p_l2c2 = (u16*)alloc(128 * 128 * 2);
  u16* p_c2l1 = (u16*)alloc(128 * 128 * 2);
  u16* p_c2l2 = (u16*)alloc(128 * 128 * 2);
  u16* p_l2l1 = (u16*)alloc(128 * 128 * 2);
  u16* p_l2l2 = (u16*)alloc(128 * 128 * 2);
  u16* p_cu   = (u16*)alloc(256 * 128 * 2);
  u16* p_lu   = (u16*)alloc(384 * 128 * 2);

  // Degrees + per-edge normalization
  (void)hipMemsetAsync(ldeg, 0, (size_t)L * 4, stream);
  (void)hipMemsetAsync(cdeg, 0, (size_t)C * 4, stream);
  deg_kernel<<<(E + 255) / 256, 256, 0, stream>>>(l_idx, c_idx, ldeg, cdeg, E);
  norm_kernel<<<(E + 255) / 256, 256, 0, stream>>>(l_idx, c_idx, ldeg, cdeg, invn, E);

  // Pack all weight matrices into WMMA B-fragment layout (bf16)
  pack_w_kernel<<<8 * 4,  32, 0, stream>>>(w_l2c1, p_l2c1, 4);
  pack_w_kernel<<<8 * 4,  32, 0, stream>>>(w_l2c2, p_l2c2, 4);
  pack_w_kernel<<<8 * 4,  32, 0, stream>>>(w_c2l1, p_c2l1, 4);
  pack_w_kernel<<<8 * 4,  32, 0, stream>>>(w_c2l2, p_c2l2, 4);
  pack_w_kernel<<<8 * 4,  32, 0, stream>>>(w_l2l1, p_l2l1, 4);
  pack_w_kernel<<<8 * 4,  32, 0, stream>>>(w_l2l2, p_l2l2, 4);
  pack_w_kernel<<<8 * 8,  32, 0, stream>>>(w_cu,   p_cu,   8);
  pack_w_kernel<<<8 * 12, 32, 0, stream>>>(w_lu,   p_lu,  12);

  // Tile initial embeddings
  init_emb_kernel<<<(L * 128) / 256, 256, 0, stream>>>(l_init, l_emb, L * 128);
  init_emb_kernel<<<(C * 128) / 256, 256, 0, stream>>>(c_init, c_emb, C * 128);

  for (int it = 0; it < 4; ++it) {
    // Messages from current embeddings (before any in-place update)
    mlp2_kernel<<<L / TS, 128, 0, stream>>>(l_emb, 0, p_l2c1, b_l2c1, p_l2c2, b_l2c2, msg_l2c);
    mlp2_kernel<<<L / TS, 128, 0, stream>>>(l_emb, 1, p_l2l1, b_l2l1, p_l2l2, b_l2l2, l2lmsg);
    mlp2_kernel<<<C / TS, 128, 0, stream>>>(c_emb, 0, p_c2l1, b_c2l1, p_c2l2, b_c2l2, msg_c2l);

    // Normalized segment sums
    (void)hipMemsetAsync(aggr_c, 0, (size_t)C * 128 * 4, stream);
    scatter_kernel<<<(E + 7) / 8, 256, 0, stream>>>(msg_l2c, l_idx, c_idx, invn, aggr_c, E);
    (void)hipMemsetAsync(aggr_l, 0, (size_t)L * 128 * 4, stream);
    scatter_kernel<<<(E + 7) / 8, 256, 0, stream>>>(msg_c2l, c_idx, l_idx, invn, aggr_l, E);

    // In-place node updates
    update_kernel<2><<<C / TS, 128, 0, stream>>>(c_emb, aggr_c, nullptr, p_cu, b_cu, c_emb);
    update_kernel<3><<<L / TS, 128, 0, stream>>>(l_emb, aggr_l, l2lmsg, p_lu, b_lu, l_emb);
  }
}
